// MultiHeadAttention_63230508531947
// MI455X (gfx1250) — matrix operand order repaired and checked by reading, not compiled, not run on hardware
//
#include <hip/hip_runtime.h>
#include <hip/hip_bf16.h>
#include <math.h>

// ---------------------------------------------------------------------------
// MultiHeadAttention (B=4, S=2048, D_MODEL=512, H=8, Dh=64) for gfx1250.
// bf16 WMMA (v_wmma_f32_16x16x32_bf16) everywhere, f32 softmax/LN epilogues.
// ---------------------------------------------------------------------------

typedef __bf16 bf16_t;
typedef __attribute__((ext_vector_type(16))) __bf16 v16bf;
typedef __attribute__((ext_vector_type(8)))  float  v8f;

union FragU {
    v16bf  v;
    uint4  q[2];
    bf16_t e[16];
};

static constexpr int BATCH = 4;
static constexpr int SEQ   = 2048;
static constexpr int DM    = 512;
static constexpr int NH    = 8;
static constexpr int DH    = 64;
static constexpr int MTOT  = BATCH * SEQ;      // 8192 rows

// ---- WMMA wrapper ----------------------------------------------------------
__device__ inline v8f wmma_bf16(v16bf a, v16bf b, v8f c) {
    return __builtin_amdgcn_wmma_f32_16x16x32_bf16(
        /*neg_a=*/false, a, /*neg_b=*/false, b,
        /*c_mod=*/(short)0, c, /*reuse_a=*/false, /*reuse_b=*/false);
}

// A fragment (16x32, MxK): lane = m(0..15), hi-half lanes take K base +8;
// per lane: K kb..kb+7 in v[0..3], K kb+16..kb+23 in v[4..7].
__device__ inline v16bf load_a_frag(const bf16_t* A, int row0, int ld, int k0, int lane) {
    int m  = lane & 15;
    int kb = k0 + ((lane >> 4) << 3);
    const bf16_t* p = A + (size_t)(row0 + m) * ld + kb;
    FragU f;
    f.q[0] = *(const uint4*)(p);
    f.q[1] = *(const uint4*)(p + 16);
    return f.v;
}

// B fragment (32x16, KxN) from B^T storage Bt[n][k] (ld = K stride):
// lane = n(0..15), hi-half lanes take K base +8; K kb..kb+7 and kb+16..kb+23 per lane.
__device__ inline v16bf load_b_frag(const bf16_t* Bt, int col0, int ld, int k0, int lane) {
    int n  = lane & 15;
    int kb = k0 + ((lane >> 4) << 3);
    const bf16_t* p = Bt + (size_t)(col0 + n) * ld + kb;
    FragU f;
    f.q[0] = *(const uint4*)(p);
    f.q[1] = *(const uint4*)(p + 16);
    return f.v;
}

// A fragment built from an f32 row-major strip (LDS probabilities), cvt->bf16.
__device__ inline v16bf load_a_frag_f32(const float* A, int ld, int k0, int lane) {
    int m  = lane & 15;
    int kb = k0 + ((lane >> 4) << 3);
    const float* p = A + (size_t)m * ld + kb;
    float4 x0 = *(const float4*)(p);
    float4 x1 = *(const float4*)(p + 4);
    float4 y0 = *(const float4*)(p + 16);
    float4 y1 = *(const float4*)(p + 20);
    FragU f;
    f.e[0]=(bf16_t)x0.x; f.e[1]=(bf16_t)x0.y; f.e[2]=(bf16_t)x0.z; f.e[3]=(bf16_t)x0.w;
    f.e[4]=(bf16_t)x1.x; f.e[5]=(bf16_t)x1.y; f.e[6]=(bf16_t)x1.z; f.e[7]=(bf16_t)x1.w;
    f.e[8]=(bf16_t)y0.x; f.e[9]=(bf16_t)y0.y; f.e[10]=(bf16_t)y0.z; f.e[11]=(bf16_t)y0.w;
    f.e[12]=(bf16_t)y1.x; f.e[13]=(bf16_t)y1.y; f.e[14]=(bf16_t)y1.z; f.e[15]=(bf16_t)y1.w;
    return f.v;
}

// ---------------------------------------------------------------------------
// Prep kernels: f32 -> bf16 copies, and 512x512 weight transpose to bf16.
// ---------------------------------------------------------------------------
__global__ __launch_bounds__(256)
void cvt_bf16_kernel(const float* __restrict__ src, bf16_t* __restrict__ dst, int n) {
    int stride = gridDim.x * blockDim.x;
    for (int i = blockIdx.x * blockDim.x + threadIdx.x; i < n; i += stride)
        dst[i] = (bf16_t)src[i];
}

__global__ __launch_bounds__(256)
void transpose_w_kernel(const float* __restrict__ W, bf16_t* __restrict__ Wt) {
    int stride = gridDim.x * blockDim.x;
    for (int i = blockIdx.x * blockDim.x + threadIdx.x; i < DM * DM; i += stride) {
        int n = i >> 9, k = i & 511;
        Wt[i] = (bf16_t)W[k * DM + n];       // Wt[n][k] = W[k][n]
    }
}

// ---------------------------------------------------------------------------
// Fused Q/K/V projection GEMM. grid = (MTOT/16, 3); blockIdx.y selects the
// input/weight pair. One 16-row tile per block; 8 waves each own 4 column
// tiles held as live accumulators (k outer loop -> A loaded once per k-step,
// no operand hoisting/spilling).
//   sel 0: Q -> bf16 [B,H,S,Dh]   sel 1: K -> bf16 [B,H,S,Dh]
//   sel 2: V -> bf16 [B,H,Dh,S] (transposed for the P@V B operand)
// ---------------------------------------------------------------------------
__global__ __launch_bounds__(256)
void gemm_qkv_kernel(const bf16_t* __restrict__ qbf,
                     const bf16_t* __restrict__ kbf,
                     const bf16_t* __restrict__ vbf,
                     const bf16_t* __restrict__ WQt,
                     const bf16_t* __restrict__ WKt,
                     const bf16_t* __restrict__ WVt,
                     bf16_t* __restrict__ Qh,
                     bf16_t* __restrict__ Kh,
                     bf16_t* __restrict__ Vt) {
    const int sel = blockIdx.y;
    const bf16_t* A  = (sel == 0) ? qbf : (sel == 1) ? kbf : vbf;
    const bf16_t* Bt = (sel == 0) ? WQt : (sel == 1) ? WKt : WVt;
    bf16_t* dst      = (sel == 0) ? Qh  : (sel == 1) ? Kh  : Vt;

    const int m0   = blockIdx.x * 16;
    const int wave = threadIdx.x >> 5;
    const int lane = threadIdx.x & 31;
    const int mloc = (lane >> 4) << 3;     // 0 or 8
    const int nloc = lane & 15;

    v8f acc[4] = {{}, {}, {}, {}};
#pragma unroll 2
    for (int k0 = 0; k0 < DM; k0 += 32) {
        v16bf a = load_a_frag(A, m0, DM, k0, lane);
#pragma unroll
        for (int i = 0; i < 4; ++i) {
            int n0 = (wave + 8 * i) * 16;
            acc[i] = wmma_bf16(a, load_b_frag(Bt, n0, DM, k0, lane), acc[i]);
        }
    }

#pragma unroll
    for (int i = 0; i < 4; ++i) {
        int n0 = (wave + 8 * i) * 16;
#pragma unroll
        for (int r = 0; r < 8; ++r) {
            int m = m0 + r + mloc;         // global row (b*SEQ + s)
            int n = n0 + nloc;             // global col
            int b = m >> 11, s = m & (SEQ - 1);
            int h = n >> 6,  d = n & (DH - 1);
            if (sel == 2) {                // V transposed: [b,h,d,s]
                dst[(((size_t)(b * NH + h)) * DH + d) * SEQ + s] = (bf16_t)acc[i][r];
            } else {                       // Q/K: [b,h,s,d]
                dst[(((size_t)(b * NH + h)) * SEQ + s) * DH + d] = (bf16_t)acc[i][r];
            }
        }
    }
}

// ---------------------------------------------------------------------------
// Output projection GEMM: pre[M x 512] = ctx @ WOt^T + residual (f32).
// Same register-resident multi-accumulator structure.
// ---------------------------------------------------------------------------
__global__ __launch_bounds__(256)
void gemm_out_kernel(const bf16_t* __restrict__ A,
                     const bf16_t* __restrict__ Bt,
                     const float*  __restrict__ resid,
                     float* __restrict__ out_f32) {
    const int m0   = blockIdx.x * 16;
    const int wave = threadIdx.x >> 5;
    const int lane = threadIdx.x & 31;
    const int mloc = (lane >> 4) << 3;
    const int nloc = lane & 15;

    v8f acc[4] = {{}, {}, {}, {}};
#pragma unroll 2
    for (int k0 = 0; k0 < DM; k0 += 32) {
        v16bf a = load_a_frag(A, m0, DM, k0, lane);
#pragma unroll
        for (int i = 0; i < 4; ++i) {
            int n0 = (wave + 8 * i) * 16;
            acc[i] = wmma_bf16(a, load_b_frag(Bt, n0, DM, k0, lane), acc[i]);
        }
    }

#pragma unroll
    for (int i = 0; i < 4; ++i) {
        int n0 = (wave + 8 * i) * 16;
#pragma unroll
        for (int r = 0; r < 8; ++r) {
            int m = m0 + r + mloc;
            int n = n0 + nloc;
            out_f32[(size_t)m * DM + n] = acc[i][r] + resid[(size_t)m * DM + n];
        }
    }
}

// ---------------------------------------------------------------------------
// Attention: one block = one (b,h) and 16 query rows.
//   Phase 1: scores 16x2048 via WMMA -> LDS (f32, scaled).
//   Phase 2: mask + f32 softmax in LDS (mask read as contiguous bytes),
//            probabilities written once to d_out.
//   Phase 3: context = P @ V via WMMA reading P from LDS (split-K over waves).
// LDS strip: 16*2048 f32 = 128 KB (fits CDNA5's 320 KB WGP LDS).
// ---------------------------------------------------------------------------
static constexpr int ATTN_SMEM_FLOATS = 16 * SEQ + 256 + 32 + 4 * 256;
static constexpr int ATTN_SMEM_BYTES  = ATTN_SMEM_FLOATS * 4;   // 136320 B

__global__ __launch_bounds__(256)
void attention_kernel(const bf16_t* __restrict__ Qh,
                      const bf16_t* __restrict__ Kh,
                      const bf16_t* __restrict__ Vt,
                      const unsigned char* __restrict__ mask,
                      float*  __restrict__ attn_out,
                      bf16_t* __restrict__ ctx) {
    extern __shared__ float smem[];
    float* Ps      = smem;                 // [16][2048]
    float* red     = Ps + 16 * SEQ;        // [16][16]
    float* rowstat = red + 256;            // rowmax[16], rowsum[16]
    float* pc      = rowstat + 32;         // [4][16][16] split-K partials

    const int q0 = blockIdx.x * 16;
    const int h  = blockIdx.y;
    const int b  = blockIdx.z;
    const int wave = threadIdx.x >> 5;
    const int lane = threadIdx.x & 31;
    const int mloc = (lane >> 4) << 3;
    const int nloc = lane & 15;

    const size_t headQK = ((size_t)(b * NH + h)) * SEQ * DH;
    const bf16_t* Qb = Qh + headQK + (size_t)q0 * DH;
    const bf16_t* Kb = Kh + headQK;
    const bf16_t* Vb = Vt + headQK;        // [DH][SEQ]

    // ---- Phase 1: scores -> LDS (pure WMMA stream, no mask here) ----
    v16bf aq0 = load_a_frag(Qb, 0, DH, 0,  lane);
    v16bf aq1 = load_a_frag(Qb, 0, DH, 32, lane);
    for (int t = wave; t < SEQ / 16; t += 8) {
        int key0 = t * 16;
        if (t + 8 < SEQ / 16)              // hide HBM latency behind WMMA
            __builtin_prefetch(Kb + (size_t)(key0 + 128) * DH, 0, 1);
        v8f c = {};
        c = wmma_bf16(aq0, load_b_frag(Kb, key0, DH, 0,  lane), c);
        c = wmma_bf16(aq1, load_b_frag(Kb, key0, DH, 32, lane), c);
#pragma unroll
        for (int r = 0; r < 8; ++r) {
            int m   = r + mloc;
            int key = key0 + nloc;
            Ps[m * SEQ + key] = c[r] * 0.125f;   // 1/sqrt(64)
        }
    }
    __syncthreads();

    // ---- Phase 2: mask + softmax rows (16 threads per row) ----
    const int g = threadIdx.x >> 4;        // row 0..15
    const int j = threadIdx.x & 15;
    float* row = Ps + g * SEQ;
    const unsigned char* mrow = mask + ((size_t)b * SEQ + (q0 + g)) * SEQ;

    float mx = -INFINITY;
    for (int cidx = j; cidx < SEQ; cidx += 16) {
        float sv = mrow[cidx] ? -INFINITY : row[cidx];
        row[cidx] = sv;
        mx = fmaxf(mx, sv);
    }
    red[g * 16 + j] = mx;
    __syncthreads();
    if (j == 0) {
        float mm = red[g * 16];
        for (int t = 1; t < 16; ++t) mm = fmaxf(mm, red[g * 16 + t]);
        rowstat[g] = mm;
    }
    __syncthreads();
    const float rmax = rowstat[g];

    float sacc = 0.0f;
    for (int cidx = j; cidx < SEQ; cidx += 16) {
        float e = __expf(row[cidx] - rmax);
        row[cidx] = e;
        sacc += e;
    }
    red[g * 16 + j] = sacc;
    __syncthreads();
    if (j == 0) {
        float ss = 0.0f;
        for (int t = 0; t < 16; ++t) ss += red[g * 16 + t];
        rowstat[16 + g] = ss;
    }
    __syncthreads();
    const float inv = 1.0f / rowstat[16 + g];

    float* attn_row = attn_out + (((size_t)(b * NH + h)) * SEQ + (q0 + g)) * SEQ;
    for (int cidx = j; cidx < SEQ; cidx += 16) {
        float p = row[cidx] * inv;
        row[cidx] = p;
        attn_row[cidx] = p;                // the mandatory 537 MB stream
    }
    __syncthreads();

    // ---- Phase 3: context = P @ V  (split-K: waves 0-3 low half, 4-7 high) ----
    const int dt   = wave & 3;             // d tile: 16 cols of 64
    const int half = wave >> 2;
    v8f c = {};
    const int kbeg = half * (SEQ / 2);
#pragma unroll 2
    for (int kk = 0; kk < SEQ / 2; kk += 32) {
        v16bf a   = load_a_frag_f32(Ps, SEQ, kbeg + kk, lane);
        v16bf bfr = load_b_frag(Vb, dt * 16, SEQ, kbeg + kk, lane);
        c = wmma_bf16(a, bfr, c);
    }
    if (half == 1) {
#pragma unroll
        for (int r = 0; r < 8; ++r)
            pc[dt * 256 + (r + mloc) * 16 + nloc] = c[r];
    }
    __syncthreads();
    if (half == 0) {
#pragma unroll
        for (int r = 0; r < 8; ++r) {
            int m = r + mloc;
            float v = c[r] + pc[dt * 256 + m * 16 + nloc];
            size_t rowi = (size_t)(b * SEQ + q0 + m) * DM + h * DH + dt * 16 + nloc;
            ctx[rowi] = (bf16_t)v;
        }
    }
}

// ---------------------------------------------------------------------------
// LayerNorm over 512-wide rows (residual already folded in by gemm_out).
// ---------------------------------------------------------------------------
__global__ __launch_bounds__(256)
void layernorm_kernel(const float* __restrict__ pre,
                      const float* __restrict__ gamma,
                      const float* __restrict__ beta,
                      float* __restrict__ out) {
    __shared__ float sb[256];
    const int row = blockIdx.x;
    const int t   = threadIdx.x;
    const float* pr = pre + (size_t)row * DM;

    float a = pr[t], b2 = pr[t + 256];
    sb[t] = a + b2;
    __syncthreads();
    for (int off = 128; off > 0; off >>= 1) {
        if (t < off) sb[t] += sb[t + off];
        __syncthreads();
    }
    const float mu = sb[0] * (1.0f / DM);
    __syncthreads();

    float d0 = a - mu, d1 = b2 - mu;
    sb[t] = d0 * d0 + d1 * d1;
    __syncthreads();
    for (int off = 128; off > 0; off >>= 1) {
        if (t < off) sb[t] += sb[t + off];
        __syncthreads();
    }
    const float invstd = rsqrtf(sb[0] * (1.0f / DM) + 1e-5f);

    float* orow = out + (size_t)row * DM;
    orow[t]       = d0 * invstd * gamma[t]       + beta[t];
    orow[t + 256] = d1 * invstd * gamma[t + 256] + beta[t + 256];
}

// ---------------------------------------------------------------------------
// Host launcher
// ---------------------------------------------------------------------------
extern "C" void kernel_launch(void* const* d_in, const int* in_sizes, int n_in,
                              void* d_out, int out_size, void* d_ws, size_t ws_size,
                              hipStream_t stream) {
    const float* query = (const float*)d_in[0];
    const float* key   = (const float*)d_in[1];
    const float* value = (const float*)d_in[2];
    const unsigned char* mask = (const unsigned char*)d_in[3];
    const float* W_Q = (const float*)d_in[4];
    const float* W_K = (const float*)d_in[5];
    const float* W_V = (const float*)d_in[6];
    const float* W_O = (const float*)d_in[7];
    const float* ln_gamma = (const float*)d_in[8];
    const float* ln_beta  = (const float*)d_in[9];

    // d_out: [output f32 4*2048*512][attn_score f32 4*8*2048*2048]
    float* out_final = (float*)d_out;
    float* attn_out  = out_final + (size_t)MTOT * DM;

    // workspace carve-up (bytes)
    char* W = (char*)d_ws;
    const size_t SZ_INBF = (size_t)MTOT * DM * 2;   // 8 MB
    const size_t SZ_WT   = (size_t)DM * DM * 2;     // 0.5 MB
    bf16_t* qbf = (bf16_t*)(W);
    bf16_t* kbf = (bf16_t*)(W + SZ_INBF);
    bf16_t* vbf = (bf16_t*)(W + 2 * SZ_INBF);
    bf16_t* WQt = (bf16_t*)(W + 3 * SZ_INBF);
    bf16_t* WKt = (bf16_t*)(W + 3 * SZ_INBF + SZ_WT);
    bf16_t* WVt = (bf16_t*)(W + 3 * SZ_INBF + 2 * SZ_WT);
    bf16_t* WOt = (bf16_t*)(W + 3 * SZ_INBF + 3 * SZ_WT);
    char*   p   = W + 3 * SZ_INBF + 4 * SZ_WT;
    bf16_t* Qh  = (bf16_t*)(p);                 // [B,H,S,Dh] bf16
    bf16_t* Kh  = (bf16_t*)(p + SZ_INBF);       // [B,H,S,Dh] bf16
    bf16_t* Vt  = (bf16_t*)(p + 2 * SZ_INBF);   // [B,H,Dh,S] bf16
    bf16_t* ctx = (bf16_t*)(p + 3 * SZ_INBF);   // [M, 512]  bf16
    float*  pre = (float*)(p + 4 * SZ_INBF);    // [M, 512]  f32

    const int NTOK = MTOT * DM;                 // 4.19M

    // bf16 conversions + weight transposes
    cvt_bf16_kernel<<<2048, 256, 0, stream>>>(query, qbf, NTOK);
    cvt_bf16_kernel<<<2048, 256, 0, stream>>>(key,   kbf, NTOK);
    cvt_bf16_kernel<<<2048, 256, 0, stream>>>(value, vbf, NTOK);
    transpose_w_kernel<<<512, 256, 0, stream>>>(W_Q, WQt);
    transpose_w_kernel<<<512, 256, 0, stream>>>(W_K, WKt);
    transpose_w_kernel<<<512, 256, 0, stream>>>(W_V, WVt);
    transpose_w_kernel<<<512, 256, 0, stream>>>(W_O, WOt);

    // fused Q/K/V projections (WMMA)
    dim3 qkv_grid(MTOT / 16, 3);
    gemm_qkv_kernel<<<qkv_grid, 256, 0, stream>>>(qbf, kbf, vbf,
                                                  WQt, WKt, WVt,
                                                  Qh, Kh, Vt);

    // attention core (WMMA + LDS-resident softmax strip)
    dim3 agrid(SEQ / 16, NH, BATCH);
    attention_kernel<<<agrid, 256, ATTN_SMEM_BYTES, stream>>>(Qh, Kh, Vt, mask,
                                                              attn_out, ctx);

    // output projection + residual (WMMA), then LayerNorm
    gemm_out_kernel<<<MTOT / 16, 256, 0, stream>>>(ctx, WOt, query, pre);
    layernorm_kernel<<<MTOT, 256, 0, stream>>>(pre, ln_gamma, ln_beta, out_final);

    (void)in_sizes; (void)n_in; (void)out_size; (void)ws_size;
}
